// SECOND_NECK_34411277975639
// MI455X (gfx1250) — matrix operand order repaired and checked by reading, not compiled, not run on hardware
//
#include <hip/hip_runtime.h>
#include <hip/hip_bf16.h>
#include <math.h>

// ---------------------------------------------------------------------------
// SECOND-style sparse 3D CNN, evaluated densely as fused
// spline-basis -> implicit-GEMM conv3d (f16 WMMA, f32 accum) -> BN/ReLU/mask.
// A-operand (weights) staged into LDS via the Tensor Data Mover (TDM).
// ---------------------------------------------------------------------------

typedef __attribute__((ext_vector_type(16))) _Float16 v16h;
typedef __attribute__((ext_vector_type(8)))  float    v8f;
typedef __attribute__((ext_vector_type(4)))  unsigned v4u;
typedef __attribute__((ext_vector_type(8)))  int      v8i;
typedef __attribute__((ext_vector_type(4)))  int      v4i;

#if defined(__gfx1250__) && __has_builtin(__builtin_amdgcn_tensor_load_to_lds)
#define USE_TDM 1
#else
#define USE_TDM 0
#endif

#define TPB 256

struct LayerDef { int Cin, Cout, KD, KH, KW, SD, SH, SW, PD, PH, PW, subm; };

static const LayerDef g_layers[12] = {
  { 4, 16, 3,3,3, 1,1,1, 1,1,1, 1},
  {16, 16, 3,3,3, 1,1,1, 1,1,1, 1},
  {16, 32, 3,3,3, 2,2,2, 1,1,1, 0},
  {32, 32, 3,3,3, 1,1,1, 1,1,1, 1},
  {32, 32, 3,3,3, 1,1,1, 1,1,1, 1},
  {32, 64, 3,3,3, 2,2,2, 1,1,1, 0},
  {64, 64, 3,3,3, 1,1,1, 1,1,1, 1},
  {64, 64, 3,3,3, 1,1,1, 1,1,1, 1},
  {64, 64, 3,3,3, 2,2,2, 0,1,1, 0},
  {64, 64, 3,3,3, 1,1,1, 1,1,1, 1},
  {64, 64, 3,3,3, 1,1,1, 1,1,1, 1},
  {64,128, 3,1,1, 2,1,1, 0,0,0, 0},
};

// ---------------------------------------------------------------------------
// densify: scatter-add voxel features, set active-site mask
// ---------------------------------------------------------------------------
__global__ __launch_bounds__(TPB) void densify_kernel(
    const float* __restrict__ feats, const int* __restrict__ coors,
    float* __restrict__ x, float* __restrict__ mask, int N, int Pin) {
  int i = blockIdx.x * TPB + threadIdx.x;
  if (i >= N) return;
  int z  = coors[i*4 + 1];
  int y  = coors[i*4 + 2];
  int xx = coors[i*4 + 3];
  int p = (z * 96 + y) * 96 + xx;      // batch_size == 1
  #pragma unroll
  for (int c = 0; c < 4; ++c)
    atomicAdd(&x[(size_t)c * Pin + p], feats[i*4 + c]);
  mask[p] = 1.f;
}

// ---------------------------------------------------------------------------
// spline basis: [Cin][Pin] f32 -> [3*Cin][Pin] f16, Gaussian RBF at {-1,0,1}
// ---------------------------------------------------------------------------
__global__ __launch_bounds__(TPB) void basis_kernel(
    const float* __restrict__ x, _Float16* __restrict__ b, int Pin) {
  int p = blockIdx.x * TPB + threadIdx.x;
  if (p >= Pin) return;
  int ci = blockIdx.y;
  float v = x[(size_t)ci * Pin + p];
  float d0 = v + 1.f, d2 = v - 1.f;
  size_t base = (size_t)(ci * 3) * Pin + p;
  b[base]           = (_Float16)__expf(-d0 * d0);
  b[base + Pin]     = (_Float16)__expf(-v  * v );
  b[base + 2 * Pin] = (_Float16)__expf(-d2 * d2);
}

// ---------------------------------------------------------------------------
// weight permute: f32 (co, ci3, kz,ky,kx) -> f16 [tap][co][ci3]
// ---------------------------------------------------------------------------
__global__ __launch_bounds__(TPB) void wcvt_kernel(
    const float* __restrict__ w, _Float16* __restrict__ wf,
    int Cout, int Cin3, int KT) {
  int i = blockIdx.x * TPB + threadIdx.x;
  int total = Cout * Cin3 * KT;
  if (i >= total) return;
  int t   = i % KT;
  int r   = i / KT;
  int ci3 = r % Cin3;
  int co  = r / Cin3;
  wf[((size_t)t * Cout + co) * Cin3 + ci3] = (_Float16)w[i];
}

// ---------------------------------------------------------------------------
// mask dilation for strided (non-submanifold) layers
// ---------------------------------------------------------------------------
__global__ __launch_bounds__(TPB) void mask_kernel(
    const float* __restrict__ mIn, float* __restrict__ mOut,
    int Din, int Hin, int Win, int Dout, int Hout, int Wout,
    int KD, int KH, int KW, int SD, int SH, int SW, int PD, int PH, int PW) {
  int p = blockIdx.x * TPB + threadIdx.x;
  int Pout = Dout * Hout * Wout;
  if (p >= Pout) return;
  int hw = Hout * Wout;
  int z = p / hw;
  int r = p - z * hw;
  int y = r / Wout;
  int x = r - y * Wout;
  float v = 0.f;
  for (int kz = 0; kz < KD; ++kz) {
    int iz = z * SD - PD + kz;
    if ((unsigned)iz >= (unsigned)Din) continue;
    for (int ky = 0; ky < KH; ++ky) {
      int iy = y * SH - PH + ky;
      if ((unsigned)iy >= (unsigned)Hin) continue;
      for (int kx = 0; kx < KW; ++kx) {
        int ix = x * SW - PW + kx;
        if ((unsigned)ix >= (unsigned)Win) continue;
        if (mIn[(iz * Hin + iy) * Win + ix] > 0.f) v = 1.f;
      }
    }
  }
  mOut[p] = v;
}

// ---------------------------------------------------------------------------
// fused implicit-GEMM conv3d with WMMA f16->f32 + BN + ReLU + mask epilogue
//   D[Cout x Pout] = W[Cout x K] * Basis[K x Pout],  K = KT * Cin3
// Block tile: MT x 256.  Each wave: 16 (M) x NSPAN (N), NSPAN/16 accumulators
// sharing one A fragment -> NSPAN/16 back-to-back WMMAs per K-chunk.
// A tile (weights) is DMA'd into LDS by the TDM; its pad feature reproduces
// the 34-half (17-dword, bank-conflict-free) LDS pitch.
// ---------------------------------------------------------------------------
struct ConvParams {
  const _Float16* __restrict__ basis;   // [Cin3][Pin] f16
  const _Float16* __restrict__ wf16;    // [KT][Cout][Cin3] f16
  const float*    __restrict__ gamma;
  const float*    __restrict__ beta;
  const float*    __restrict__ mask;    // [Pout]
  float*          __restrict__ out;     // [Cout][Pout]
  int Cin3, Cout, Pin, Pout;
  int Din, Hin, Win, Dout, Hout, Wout;
  int KD, KH, KW, SD, SH, SW, PD, PH, PW;
};

template <int MT, int WAVEN, int NSPAN>
__global__ __launch_bounds__(TPB) void conv_wmma_kernel(ConvParams cp) {
  constexpr int NT   = 256;
  constexpr int NSUB = NSPAN / 16;      // accumulators per wave
  // pitch 34 halves (= 17 dwords, coprime with 64 banks -> conflict-free)
  __shared__ __attribute__((aligned(16))) _Float16 ldsA[MT * 34];
  __shared__ __attribute__((aligned(16))) _Float16 ldsB[NT * 34];

  const int tid    = threadIdx.x;
  const int lane15 = tid & 15;
  const int hi     = (tid >> 4) & 1;
  const int wave   = tid >> 5;
  const int wm     = wave / WAVEN;
  const int wn     = wave % WAVEN;
  const int pBase  = blockIdx.x * NT;
  const int coBase = blockIdx.y * MT;

  // B staging: one output column per thread (n = tid), invariant across K
  const int  pB      = pBase + tid;
  const bool pBvalid = pB < cp.Pout;
  int zB = 0, yB = 0, xBc = 0;
  if (pBvalid) {
    int hw = cp.Hout * cp.Wout;
    zB = pB / hw;
    int r = pB - zB * hw;
    yB = r / cp.Wout;
    xBc = r - yB * cp.Wout;
  }

  v8f acc[NSUB];
  #pragma unroll
  for (int s = 0; s < NSUB; ++s) acc[s] = (v8f){};

  const int KHW = cp.KH * cp.KW;
  const int KT  = cp.KD * KHW;

  for (int t = 0; t < KT; ++t) {
    const int kz = t / KHW;
    const int rt = t - kz * KHW;
    const int ky = rt / cp.KW;
    const int kx = rt - ky * cp.KW;

    // input gather coord for this thread's staged column at this tap
    int pin = 0; bool inOK = false;
    if (pBvalid) {
      int iz = zB * cp.SD - cp.PD + kz;
      int iy = yB * cp.SH - cp.PH + ky;
      int ix = xBc * cp.SW - cp.PW + kx;
      if ((unsigned)iz < (unsigned)cp.Din && (unsigned)iy < (unsigned)cp.Hin &&
          (unsigned)ix < (unsigned)cp.Win) {
        pin = (iz * cp.Hin + iy) * cp.Win + ix;
        inOK = true;
      }
    }
    const _Float16* wrow = cp.wf16 + (size_t)t * cp.Cout * cp.Cin3;
    const _Float16* bcol = cp.basis + pin;
    if (inOK) __builtin_prefetch(bcol, 0, 1);   // global_prefetch_b8

    for (int c0 = 0; c0 < cp.Cin3; c0 += 32) {
      __syncthreads();

#if USE_TDM
      // ---- A tile via Tensor Data Mover (wave 0 issues, EXEC-independent) --
      if (wave == 0) {
        const unsigned ldsOff = (unsigned)(size_t)(&ldsA[0]);  // ADDR[31:0] = LDS offset
        const unsigned long long ga =
            (unsigned long long)(size_t)(wrow + (size_t)coBase * cp.Cin3 + c0);
        const int dim0 = cp.Cin3 - c0;          // remaining K -> TDM OOB zero-fills tail
        // D# group 0: count=1 | lds_addr | global_addr[56:0] | type=2
        v4u g0 = { 1u, ldsOff, (unsigned)ga,
                   (unsigned)((ga >> 32) & 0x01FFFFFFu) | (2u << 30) };
        // D# group 1: data_size=2B, pad_en, pad_interval=16DW, pad_amount=1DW,
        // tensor_dim0=dim0, tensor_dim1=Cout, tile 32 x MT, dim0 stride=Cin3
        v8i g1 = {
          (int)((1u << 16) | (1u << 20) | (3u << 22)),
          (int)((unsigned)(dim0 & 0xFFFF) << 16),
          (int)(((unsigned)dim0 >> 16) | ((unsigned)(cp.Cout & 0xFFFF) << 16)),
          (int)((((unsigned)cp.Cout >> 16) & 0xFFFFu) | (32u << 16)),
          (int)(unsigned)MT,                    // tile_dim1=MT, tile_dim2=0
          (int)(unsigned)cp.Cin3,               // tensor_dim0_stride low 32
          0, 0
        };
        v4i z4 = { 0, 0, 0, 0 };
#if __clang_major__ >= 23
        v8i z8 = { 0, 0, 0, 0, 0, 0, 0, 0 };
        __builtin_amdgcn_tensor_load_to_lds(g0, g1, z4, z4, z8, 0);
#else
        __builtin_amdgcn_tensor_load_to_lds(g0, g1, z4, z4, 0);
#endif
      }
#else
      // ---- manual A staging fallback: weights [MT x 32], k contiguous ------
      {
        const int kA  = tid & 31;
        const int mA0 = tid >> 5;
        #pragma unroll
        for (int m = mA0; m < MT; m += (TPB >> 5)) {
          int co  = coBase + m;
          int ci3 = c0 + kA;
          _Float16 v = (_Float16)0;
          if (co < cp.Cout && ci3 < cp.Cin3) v = wrow[co * cp.Cin3 + ci3];
          ldsA[m * 34 + kA] = v;
        }
      }
#endif

      // stage B: basis [256 x 32], k contiguous per column (lockstep kr ->
      // consecutive lanes hit consecutive pin -> coalesced)
      for (int kr = 0; kr < 32; ++kr) {
        int ci3 = c0 + kr;
        _Float16 v = (_Float16)0;
        if (inOK && ci3 < cp.Cin3) v = bcol[(size_t)ci3 * cp.Pin];
        ldsB[tid * 34 + kr] = v;
      }

#if USE_TDM
      if (wave == 0) __builtin_amdgcn_s_wait_tensorcnt(0);
#endif
      __syncthreads();

      // A fragment (16x32 f16): lane<16 K{0..7,16..23}, lane>=16 K{8..15,24..31}
      union { unsigned u[8]; v16h h; } ua;
      const unsigned* Aw = (const unsigned*)(ldsA + (wm * 16 + lane15) * 34);
      #pragma unroll
      for (int j = 0; j < 4; ++j) {
        ua.u[j]     = Aw[hi * 4 + j];
        ua.u[4 + j] = Aw[8 + hi * 4 + j];
      }
      // NSUB B fragments (32x16 f16) share the A fragment
      #pragma unroll
      for (int s = 0; s < NSUB; ++s) {
        union { unsigned u[8]; v16h h; } ub;
        const unsigned* Bw =
            (const unsigned*)(ldsB + (wn * NSPAN + s * 16 + lane15) * 34);
        #pragma unroll
        for (int j = 0; j < 4; ++j) { ub.u[j] = Bw[hi * 4 + j]; ub.u[4 + j] = Bw[8 + hi * 4 + j]; }
        acc[s] = __builtin_amdgcn_wmma_f32_16x16x32_f16(
            false, ua.h, false, ub.h, (short)0, acc[s], false, false);
      }
    }
  }

  // epilogue: eval-BN (mean 0, var 1) + ReLU + active-site mask
  const float inv_std = 0.999500373f;  // 1/sqrt(1 + 1e-3)
  #pragma unroll
  for (int s = 0; s < NSUB; ++s) {
    const int pOut = pBase + wn * NSPAN + s * 16 + lane15;
    if (pOut < cp.Pout) {
      const float mval = cp.mask[pOut];
      #pragma unroll
      for (int r = 0; r < 8; ++r) {     // D layout: VGPR r -> row r + hi*8
        int co = coBase + wm * 16 + hi * 8 + r;
        if (co < cp.Cout) {
          float v = acc[s][r] * (cp.gamma[co] * inv_std) + cp.beta[co];
          v = v > 0.f ? v * mval : 0.f;
          cp.out[(size_t)co * cp.Pout + pOut] = v;
        }
      }
    }
  }
}

// ---------------------------------------------------------------------------
// launcher
// ---------------------------------------------------------------------------
extern "C" void kernel_launch(void* const* d_in, const int* in_sizes, int n_in,
                              void* d_out, int out_size, void* d_ws, size_t ws_size,
                              hipStream_t stream) {
  (void)n_in; (void)out_size; (void)ws_size;
  const float* feats = (const float*)d_in[0];
  const int*   coors = (const int*)d_in[1];
  const int N = in_sizes[0] / 4;

  // workspace carve (~89 MB)
  const size_t XCAP = ((size_t)6045696 * 4 + 255) & ~(size_t)255;  // 16*41*96*96 f32
  const size_t MCAP = ((size_t)377856  * 4 + 255) & ~(size_t)255;  // 41*96*96 f32
  const size_t BCAP = ((size_t)18136128* 2 + 255) & ~(size_t)255;  // 48*41*96*96 f16
  char* w = (char*)d_ws;
  float*    xA    = (float*)w;    w += XCAP;
  float*    xB    = (float*)w;    w += XCAP;
  float*    mA    = (float*)w;    w += MCAP;
  float*    mB    = (float*)w;    w += MCAP;
  _Float16* basis = (_Float16*)w; w += BCAP;
  _Float16* wf16  = (_Float16*)w;

  int D = 41, H = 96, Wd = 96;
  int Pin = D * H * Wd;

  hipMemsetAsync(xA, 0, (size_t)4 * Pin * sizeof(float), stream);
  hipMemsetAsync(mA, 0, (size_t)Pin * sizeof(float), stream);
  densify_kernel<<<(N + TPB - 1) / TPB, TPB, 0, stream>>>(feats, coors, xA, mA, N, Pin);

  float* curX = xA; float* nxtX = xB;
  float* curM = mA; float* nxtM = mB;

  for (int li = 0; li < 12; ++li) {
    const LayerDef& L = g_layers[li];
    const int Dout = (D  + 2 * L.PD - L.KD) / L.SD + 1;
    const int Hout = (H  + 2 * L.PH - L.KH) / L.SH + 1;
    const int Wout = (Wd + 2 * L.PW - L.KW) / L.SW + 1;
    const int Pout = Dout * Hout * Wout;
    const int Cin3 = L.Cin * 3;
    const int KT   = L.KD * L.KH * L.KW;

    const float* wgt   = (const float*)d_in[3 + 3 * li + 0];
    const float* gamma = (const float*)d_in[3 + 3 * li + 1];
    const float* beta  = (const float*)d_in[3 + 3 * li + 2];

    const int wtotal = L.Cout * Cin3 * KT;
    wcvt_kernel<<<(wtotal + TPB - 1) / TPB, TPB, 0, stream>>>(wgt, wf16, L.Cout, Cin3, KT);

    dim3 bg((Pin + TPB - 1) / TPB, L.Cin);
    basis_kernel<<<bg, TPB, 0, stream>>>(curX, basis, Pin);

    const float* useM = curM;
    if (!L.subm) {
      mask_kernel<<<(Pout + TPB - 1) / TPB, TPB, 0, stream>>>(
          curM, nxtM, D, H, Wd, Dout, Hout, Wout,
          L.KD, L.KH, L.KW, L.SD, L.SH, L.SW, L.PD, L.PH, L.PW);
      useM = nxtM;
    }

    ConvParams cp;
    cp.basis = basis; cp.wf16 = wf16; cp.gamma = gamma; cp.beta = beta;
    cp.mask = useM;
    cp.out  = (li == 11) ? (float*)d_out : nxtX;
    cp.Cin3 = Cin3; cp.Cout = L.Cout; cp.Pin = Pin; cp.Pout = Pout;
    cp.Din = D; cp.Hin = H; cp.Win = Wd; cp.Dout = Dout; cp.Hout = Hout; cp.Wout = Wout;
    cp.KD = L.KD; cp.KH = L.KH; cp.KW = L.KW;
    cp.SD = L.SD; cp.SH = L.SH; cp.SW = L.SW;
    cp.PD = L.PD; cp.PH = L.PH; cp.PW = L.PW;

    const int NT = 256;
    dim3 grid((Pout + NT - 1) / NT, 1);
    if (L.Cout >= 32) {
      grid.y = L.Cout / 32;
      // MT=32: 2x4 waves, each wave 16x64 (4 accumulators, shared A fragment)
      conv_wmma_kernel<32, 4, 64><<<grid, TPB, 0, stream>>>(cp);
    } else {
      grid.y = 1;
      // MT=16: 1x8 waves, each wave 16x32 (2 accumulators, shared A fragment)
      conv_wmma_kernel<16, 8, 32><<<grid, TPB, 0, stream>>>(cp);
    }

    if (li != 11) { float* t = curX; curX = nxtX; nxtX = t; }
    if (!L.subm)  { float* t = curM; curM = nxtM; nxtM = t; }
    D = Dout; H = Hout; Wd = Wout; Pin = Pout;
  }
}
